// QuantumCritic_55714315764000
// MI455X (gfx1250) — matrix operand order repaired and checked by reading, not compiled, not run on hardware
//
#include <hip/hip_runtime.h>

typedef __attribute__((ext_vector_type(16))) _Float16 v16h;
typedef __attribute__((ext_vector_type(8)))  _Float16 v8h;
typedef __attribute__((ext_vector_type(2)))  _Float16 v2h;
typedef __attribute__((ext_vector_type(8)))  float    v8f;

namespace {
constexpr int ROWS_PER_BLOCK = 64;   // 4 waves x 16-row WMMA tile
constexpr int THREADS        = 128;  // 4 wave32
constexpr int XDIM           = 80;   // 64 state + 16 action
constexpr int KPAD           = 96;   // K padded to 3 x 32
constexpr int H1             = 64;
constexpr int XH_S           = 104;  // f16 LDS row stride (halves): 208B, 16B-aligned, bank-friendly
constexpr int HS             = 65;   // f32 LDS row stride for h
}

// Assemble a v16h fragment from two contiguous 8-half (16B) runs in LDS.
__device__ __forceinline__ v16h load_frag(const _Float16* p, const int gap) {
  const v8h lo = *(const v8h*)p;
  const v8h hi = *(const v8h*)(p + gap);
  v16h r;
#pragma unroll
  for (int e = 0; e < 8; ++e) { r[e] = lo[e]; r[e + 8] = hi[e]; }
  return r;
}

// ---- 4-qubit statevector gates (bit is compile-time constant at call sites) ----
__device__ __forceinline__ void gate_rx(float* re, float* im, const int bit, float c, float s) {
#pragma unroll
  for (int i = 0; i < 16; ++i)
    if (!(i & bit)) {
      const int j = i | bit;
      const float ar = re[i], ai = im[i], br = re[j], bi = im[j];
      re[i] = c * ar + s * bi;  im[i] = c * ai - s * br;
      re[j] = c * br + s * ai;  im[j] = c * bi - s * ar;
    }
}
__device__ __forceinline__ void gate_ry(float* re, float* im, const int bit, float c, float s) {
#pragma unroll
  for (int i = 0; i < 16; ++i)
    if (!(i & bit)) {
      const int j = i | bit;
      const float ar = re[i], ai = im[i], br = re[j], bi = im[j];
      re[i] = c * ar - s * br;  im[i] = c * ai - s * bi;
      re[j] = s * ar + c * br;  im[j] = s * ai + c * bi;
    }
}
__device__ __forceinline__ void gate_rz(float* re, float* im, const int bit, float c, float s) {
#pragma unroll
  for (int i = 0; i < 16; ++i)
    if (!(i & bit)) {
      const int j = i | bit;
      const float ar = re[i], ai = im[i], br = re[j], bi = im[j];
      re[i] = c * ar + s * ai;  im[i] = c * ai - s * ar;   // * exp(-i t/2)
      re[j] = c * br - s * bi;  im[j] = c * bi + s * br;   // * exp(+i t/2)
    }
}
__device__ __forceinline__ void gate_cnot(float* re, float* im, const int cb, const int tb) {
#pragma unroll
  for (int i = 0; i < 16; ++i)
    if ((i & cb) && !(i & tb)) {
      const int j = i | tb;
      float tr = re[i]; re[i] = re[j]; re[j] = tr;
      float ti = im[i]; im[i] = im[j]; im[j] = ti;
    }
}

__global__ __launch_bounds__(THREADS)
void quantum_critic_fused(const float* __restrict__ state,
                          const float* __restrict__ action,
                          const float* __restrict__ W1, const float* __restrict__ b1,
                          const float* __restrict__ W2, const float* __restrict__ b2,
                          const float* __restrict__ qw,
                          const float* __restrict__ W3, const float* __restrict__ b3,
                          const float* __restrict__ W4, const float* __restrict__ b4,
                          float* __restrict__ out, int Btot) {
  __shared__ _Float16 lds_xh[ROWS_PER_BLOCK * XH_S];   // x rows, f16, K padded to 96 (zeros)
  __shared__ _Float16 lds_w1h[H1 * XH_S];              // W1 rows, f16, K padded to 96 (zeros)
  __shared__ float    lds_h[ROWS_PER_BLOCK * HS];
  __shared__ float    lds_qc[24];
  __shared__ float    lds_qs[24];

  const int t = threadIdx.x;
  const long rowBase = (long)blockIdx.x * ROWS_PER_BLOCK;

  // -------- Phase 0a: stage x = [state|action] rows into LDS as f16, zero-padded --------
  // 48 half-pairs per row: pairs 0..31 from state, 32..39 from action, 40..47 zeros.
  for (int idx = t; idx < ROWS_PER_BLOCK * 48; idx += THREADS) {
    const int r  = idx / 48;
    const int k2 = idx - r * 48;
    long grow = rowBase + r;
    if (grow >= Btot) grow = Btot - 1;
    float vx = 0.0f, vy = 0.0f;
    if (k2 < 32) {
      const float2 v = ((const float2*)(state + grow * 64))[k2];
      vx = v.x; vy = v.y;
    } else if (k2 < 40) {
      const float2 v = ((const float2*)(action + grow * 16))[k2 - 32];
      vx = v.x; vy = v.y;
    }
    v2h pk; pk[0] = (_Float16)vx; pk[1] = (_Float16)vy;
    *(v2h*)(lds_xh + r * XH_S + 2 * k2) = pk;
  }
  // -------- Phase 0b: stage W1 (64x80) into LDS as f16, zero-padded to K=96 --------
  for (int idx = t; idx < H1 * 48; idx += THREADS) {
    const int n  = idx / 48;
    const int k2 = idx - n * 48;
    float vx = 0.0f, vy = 0.0f;
    if (k2 < 40) {
      const float2 v = ((const float2*)(W1 + n * XDIM))[k2];
      vx = v.x; vy = v.y;
    }
    v2h pk; pk[0] = (_Float16)vx; pk[1] = (_Float16)vy;
    *(v2h*)(lds_w1h + n * XH_S + 2 * k2) = pk;
  }
  if (t < 24) {  // fixed circuit angles: cos/sin of theta/2, once per block
    const float a = 0.5f * qw[t];
    lds_qc[t] = cosf(a);
    lds_qs[t] = sinf(a);
  }
  // Prefetch next block's state tile (speculative; harmless at the grid tail).
  __builtin_prefetch((const char*)state + (rowBase + ROWS_PER_BLOCK) * 256 + (long)t * 128, 0, 1);
  __syncthreads();

  // ---------------- Phase 1: h = relu(x @ W1^T + b1) via WMMA ----------------
  const int wv   = t >> 5;
  const int lane = t & 31;
  const int g    = lane >> 4;   // lane group (K-split for A/B fragments)
  const int ml   = lane & 15;   // A: row-in-tile ; B/C: col-in-tile

  // A fragments (16-bit A 16x32): lane (g,ml), elem e -> K = s*32 + g*8 + e + (e>=8)*8
  // => two contiguous 8-half runs at +0 and +16 from base (s*32 + g*8).
  v16h afrag[3];
#pragma unroll
  for (int s = 0; s < 3; ++s)
    afrag[s] = load_frag(lds_xh + (wv * 16 + ml) * XH_S + s * 32 + g * 8, 16);

#pragma unroll
  for (int nt = 0; nt < 4; ++nt) {
    v8f acc = {};
#pragma unroll
    for (int s = 0; s < 3; ++s) {
      // B fragment (16-bit B 32x16): lane (g,ml), elem e -> K = s*32 + g*8 + e + (e>=8)*8
      // => two 8-half runs at +0 and +16; B[k][n] = W1[n][k] so W1h row (nt*16+ml) is contiguous in k.
      const v16h bfrag = load_frag(lds_w1h + (nt * 16 + ml) * XH_S + s * 32 + g * 8, 16);
      acc = __builtin_amdgcn_wmma_f32_16x16x32_f16(
          /*neg_a=*/false, afrag[s], /*neg_b=*/false, bfrag,
          /*c_mod=*/(short)0, acc, /*reuse_a=*/false, /*reuse_b=*/false);
    }
    // C/D layout: lane col = ml, reg r -> row = r + 8*g
    const float bias = b1[nt * 16 + ml];
#pragma unroll
    for (int r = 0; r < 8; ++r) {
      const int rowLoc = wv * 16 + r + 8 * g;
      lds_h[rowLoc * HS + nt * 16 + ml] = fmaxf(acc[r] + bias, 0.0f);
    }
  }
  __syncthreads();

  // ---------------- Phase 2: per-row angles -> 4-qubit VQC -> MLP head ----------------
  if (t < ROWS_PER_BLOCK) {
    const long grow = rowBase + t;
    const float* hr = lds_h + t * HS;

    // angles = h @ W2^T + b2   (W2 is (4,64) row-major; uniform addresses -> SMEM)
    float a0 = b2[0], a1 = b2[1], a2 = b2[2], a3 = b2[3];
#pragma unroll 8
    for (int n = 0; n < H1; ++n) {
      const float hv = hr[n];
      a0 += hv * W2[n];
      a1 += hv * W2[64 + n];
      a2 += hv * W2[128 + n];
      a3 += hv * W2[192 + n];
    }
    const float ang[4] = {a0, a1, a2, a3};

    // statevector |0000>, qubit q lives at bit (3-q)
    float re[16], im[16];
#pragma unroll
    for (int i = 0; i < 16; ++i) { re[i] = 0.0f; im[i] = 0.0f; }
    re[0] = 1.0f;

    // encoding RX(angles[q])
#pragma unroll
    for (int q = 0; q < 4; ++q) {
      const float ha = 0.5f * ang[q];
      gate_rx(re, im, 1 << (3 - q), cosf(ha), sinf(ha));
    }
    // 2 variational layers (fixed angles from LDS) + CNOT ring
#pragma unroll
    for (int l = 0; l < 2; ++l) {
#pragma unroll
      for (int q = 0; q < 4; ++q) {
        const int base = l * 12 + q * 3;
        gate_rx(re, im, 1 << (3 - q), lds_qc[base + 0], lds_qs[base + 0]);
        gate_ry(re, im, 1 << (3 - q), lds_qc[base + 1], lds_qs[base + 1]);
        gate_rz(re, im, 1 << (3 - q), lds_qc[base + 2], lds_qs[base + 2]);
      }
      gate_cnot(re, im, 1 << 3, 1 << 2);  // (c=0,t=1)
      gate_cnot(re, im, 1 << 2, 1 << 1);  // (1,2)
      gate_cnot(re, im, 1 << 1, 1 << 0);  // (2,3)
      gate_cnot(re, im, 1 << 0, 1 << 3);  // (3,0)
    }

    // Z expectations
    float p[16];
#pragma unroll
    for (int i = 0; i < 16; ++i) p[i] = re[i] * re[i] + im[i] * im[i];
    float ev[4];
#pragma unroll
    for (int q = 0; q < 4; ++q) {
      float acc = 0.0f;
#pragma unroll
      for (int i = 0; i < 16; ++i)
        acc += (((i >> (3 - q)) & 1) ? -p[i] : p[i]);
      ev[q] = acc;
    }

    // head: relu(ev @ W3^T + b3) @ W4^T + b4
    float result = b4[0];
#pragma unroll
    for (int m = 0; m < 32; ++m) {
      float acc = b3[m];
#pragma unroll
      for (int q = 0; q < 4; ++q) acc += ev[q] * W3[m * 4 + q];
      result += fmaxf(acc, 0.0f) * W4[m];
    }
    if (grow < Btot) out[grow] = result;
  }
}

extern "C" void kernel_launch(void* const* d_in, const int* in_sizes, int n_in,
                              void* d_out, int out_size, void* d_ws, size_t ws_size,
                              hipStream_t stream) {
  const float* state  = (const float*)d_in[0];
  const float* action = (const float*)d_in[1];
  const float* W1     = (const float*)d_in[2];
  const float* b1     = (const float*)d_in[3];
  const float* W2     = (const float*)d_in[4];
  const float* b2     = (const float*)d_in[5];
  const float* qw     = (const float*)d_in[6];
  const float* W3     = (const float*)d_in[7];
  const float* b3     = (const float*)d_in[8];
  const float* W4     = (const float*)d_in[9];
  const float* b4     = (const float*)d_in[10];
  float* out = (float*)d_out;

  const int B = in_sizes[0] / 64;
  const int grid = (B + ROWS_PER_BLOCK - 1) / ROWS_PER_BLOCK;
  quantum_critic_fused<<<grid, THREADS, 0, stream>>>(
      state, action, W1, b1, W2, b2, qw, W3, b3, W4, b4, out, B);
}
